// Affinity_Module_46385646797197
// MI455X (gfx1250) — matrix operand order repaired and checked by reading, not compiled, not run on hardware
//
#include <hip/hip_runtime.h>

typedef __attribute__((ext_vector_type(16))) _Float16 v16h;
typedef __attribute__((ext_vector_type(8)))  _Float16 v8h;
typedef __attribute__((ext_vector_type(8)))  float    v8f;
typedef __attribute__((ext_vector_type(4)))  float    v4f;

#define BATCH 2
#define CH    512
#define NPIX  9216          // 96*96
#define NTILE 576           // NPIX / 16
#define JSPLIT 4            // j-range split (occupancy); free in L2 traffic
#define JT_PER (NTILE / JSPLIT)   // 144 j tiles per block
#define EPSF  1e-12f

// ---------------------------------------------------------------------------
// Pass 1: L2-normalize over channels, write transposed f16 key: keyT[b][n][c]
// ---------------------------------------------------------------------------
__global__ __launch_bounds__(256) void norm_kernel(const float* __restrict__ x,
                                                   _Float16* __restrict__ keyT) {
  int idx = blockIdx.x * 256 + threadIdx.x;      // exactly BATCH*NPIX threads
  int b = idx / NPIX, n = idx % NPIX;
  const float* xp = x + (size_t)b * CH * NPIX + n;
  float s = 0.f;
  #pragma unroll 8
  for (int c = 0; c < CH; ++c) { float v = xp[(size_t)c * NPIX]; s += v * v; }
  float inv = 1.0f / fmaxf(sqrtf(s), EPSF);
  _Float16* kp = keyT + (size_t)idx * CH;
  #pragma unroll 8
  for (int c = 0; c < CH; ++c) kp[c] = (_Float16)(xp[(size_t)c * NPIX] * inv);
}

// ---------------------------------------------------------------------------
// Pass 2: sim = keyT * keyT^T per batch via WMMA f32_16x16x32_f16.
// Block = 128-row strip (8 waves x 16 rows) x one quarter of the j range.
// All waves sweep the SAME j sequence -> B fetches shared through WGP$/L0.
// A strip lives in a 16-fragment register array. Per-row L1 |sum| partials
// written per (jz, row) without atomics; summed in pass 3.
// ---------------------------------------------------------------------------
__global__ __launch_bounds__(256) void gram_kernel(const _Float16* __restrict__ keyT,
                                                   float* __restrict__ out,
                                                   float* __restrict__ l1p) {
  const int b    = blockIdx.y;
  const int jz   = blockIdx.z;              // j-range split 0..3
  const int wave = threadIdx.x >> 5;
  const int i    = blockIdx.x * 8 + wave;   // row tile 0..575, unique per wave
  const int lane = threadIdx.x & 31;
  const int sel  = lane >> 4;               // 0: lanes 0-15, 1: lanes 16-31
  const int lr   = lane & 15;

  __shared__ float sm[128];                 // 8 waves x 16 rows
  if (threadIdx.x < 128) sm[threadIdx.x] = 0.f;
  __syncthreads();

  // Preload the entire 16x512 A strip as 16 WMMA fragments (128 VGPRs).
  const _Float16* Abase =
      keyT + ((size_t)b * NPIX + (size_t)i * 16 + lr) * CH + sel * 16;
  v16h afrag[16];
  #pragma unroll
  for (int kk = 0; kk < 16; ++kk) {
    v8h a0 = *(const v8h*)(Abase + kk * 32);
    v8h a1 = *(const v8h*)(Abase + kk * 32 + 8);
    afrag[kk] = __builtin_shufflevector(a0, a1, 0, 1, 2, 3, 4, 5, 6, 7,
                                                8, 9, 10, 11, 12, 13, 14, 15);
  }

  float rs[8] = {0.f, 0.f, 0.f, 0.f, 0.f, 0.f, 0.f, 0.f};

  for (int jg = 0; jg < JT_PER / 4; ++jg) { // same j order for all waves
    const int j0 = jz * JT_PER + jg * 4;
    v8f acc0 = {}, acc1 = {}, acc2 = {}, acc3 = {};

    // B fragment base: col pixel q = j*16 + lr, K-halves start at sel*16
    const _Float16* B0 =
        keyT + ((size_t)b * NPIX + (size_t)(j0 + 0) * 16 + lr) * CH + sel * 16;
    const _Float16* B1 = B0 + (size_t)16 * CH;
    const _Float16* B2 = B1 + (size_t)16 * CH;
    const _Float16* B3 = B2 + (size_t)16 * CH;

    #pragma unroll
    for (int kk = 0; kk < 16; ++kk) {
      const int k0 = kk * 32;
      v16h bb0 = *(const v16h*)(B0 + k0);
      v16h bb1 = *(const v16h*)(B1 + k0);
      v16h bb2 = *(const v16h*)(B2 + k0);
      v16h bb3 = *(const v16h*)(B3 + k0);
      acc0 = __builtin_amdgcn_wmma_f32_16x16x32_f16(false, afrag[kk], false, bb0,
                                                    (short)0, acc0, false, false);
      acc1 = __builtin_amdgcn_wmma_f32_16x16x32_f16(false, afrag[kk], false, bb1,
                                                    (short)0, acc1, false, false);
      acc2 = __builtin_amdgcn_wmma_f32_16x16x32_f16(false, afrag[kk], false, bb2,
                                                    (short)0, acc2, false, false);
      acc3 = __builtin_amdgcn_wmma_f32_16x16x32_f16(false, afrag[kk], false, bb3,
                                                    (short)0, acc3, false, false);
    }

    // C/D layout: VGPR r holds row M = r + sel*8, column = lr
    const size_t rowbase = (size_t)b * NPIX + (size_t)i * 16 + sel * 8;
    #pragma unroll
    for (int r = 0; r < 8; ++r) {
      size_t ro = (rowbase + r) * NPIX + lr;
      float v0 = acc0[r], v1 = acc1[r], v2 = acc2[r], v3 = acc3[r];
      __builtin_nontemporal_store(v0, out + ro + (size_t)(j0 + 0) * 16);
      __builtin_nontemporal_store(v1, out + ro + (size_t)(j0 + 1) * 16);
      __builtin_nontemporal_store(v2, out + ro + (size_t)(j0 + 2) * 16);
      __builtin_nontemporal_store(v3, out + ro + (size_t)(j0 + 3) * 16);
      rs[r] += fabsf(v0) + fabsf(v1) + fabsf(v2) + fabsf(v3);
    }
  }

  // Per-wave L1 row-sum reduction across the 16 column-lanes (ds_add_f32)
  #pragma unroll
  for (int r = 0; r < 8; ++r) atomicAdd(&sm[wave * 16 + sel * 8 + r], rs[r]);
  __syncthreads();
  if (threadIdx.x < 128)
    l1p[((size_t)jz * BATCH + b) * NPIX + (size_t)blockIdx.x * 128 +
        threadIdx.x] = sm[threadIdx.x];
}

// ---------------------------------------------------------------------------
// Pass 3: out[row, :] /= max(sum_z l1p[z][row], eps), float4 streaming
// ---------------------------------------------------------------------------
__global__ __launch_bounds__(256) void scale_kernel(float* __restrict__ out,
                                                    const float* __restrict__ l1p) {
  unsigned idx = blockIdx.x * 256u + threadIdx.x;   // float4 index
  unsigned row = idx / (NPIX / 4);                  // 0 .. BATCH*NPIX-1
  float l1 = l1p[row] + l1p[(size_t)BATCH * NPIX + row] +
             l1p[(size_t)2 * BATCH * NPIX + row] +
             l1p[(size_t)3 * BATCH * NPIX + row];
  float s = 1.0f / fmaxf(l1, EPSF);
  v4f* p = (v4f*)out + idx;
  v4f v = __builtin_nontemporal_load(p);
  v *= s;
  __builtin_nontemporal_store(v, p);
}

// ---------------------------------------------------------------------------
extern "C" void kernel_launch(void* const* d_in, const int* in_sizes, int n_in,
                              void* d_out, int out_size, void* d_ws, size_t ws_size,
                              hipStream_t stream) {
  const float* x = (const float*)d_in[0];
  float* out = (float*)d_out;

  _Float16* keyT = (_Float16*)d_ws;                         // 18.9 MB
  float* l1p = (float*)((char*)d_ws +
                        (size_t)BATCH * NPIX * CH * sizeof(_Float16)); // 295 KB

  norm_kernel<<<(BATCH * NPIX) / 256, 256, 0, stream>>>(x, keyT);

  dim3 g(NPIX / 128, BATCH, JSPLIT);                        // 72 x 2 x 4 blocks
  gram_kernel<<<g, 256, 0, stream>>>(keyT, out, l1p);

  const unsigned total4 = (unsigned)((size_t)BATCH * NPIX * NPIX / 4);
  scale_kernel<<<total4 / 256, 256, 0, stream>>>(out, l1p);
}
